// InImNetDisc_42013370089556
// MI455X (gfx1250) — matrix operand order repaired and checked by reading, not compiled, not run on hardware
//
#include <hip/hip_runtime.h>
#include <hip/hip_bf16.h>

#define BATCH 8192
#define DIM   256
#define BD    (BATCH * DIM)   // 2,097,152 elements per matrix

typedef _Float16 v16h __attribute__((ext_vector_type(16)));
typedef _Float16 v8h  __attribute__((ext_vector_type(8)));
typedef float    v8f  __attribute__((ext_vector_type(8)));
typedef unsigned int v4u __attribute__((ext_vector_type(4)));
typedef int      v8i  __attribute__((ext_vector_type(8)));
typedef int      v4i  __attribute__((ext_vector_type(4)));

// ---------------------------------------------------------------------------
// TDM: 2-D tile load (rows x 256 f16, contiguous rows) into LDS.
// D# layout per CDNA5 ISA ch.8: group0 = {flags, lds_addr, global_addr, type},
// group1 = {mask/data_size, tensor dims, tile dims, strides}. 6-arg builtin
// (clang-23 / therock form); groups 2/3 + extra group zero (2-D tensor).
// ---------------------------------------------------------------------------
__device__ __forceinline__ void tdm_load_panel(unsigned lds_off, const _Float16* gptr,
                                               unsigned rows) {
  const unsigned long long ga = (unsigned long long)(uintptr_t)gptr;
  v4u g0;
  g0[0] = 1u;                                           // count=1, user descriptor
  g0[1] = lds_off;                                      // lds_addr (bytes)
  g0[2] = (unsigned)ga;                                 // global_addr[31:0]
  g0[3] = ((unsigned)(ga >> 32) & 0x01FFFFFFu) | 0x80000000u; // addr[56:32] | type=2
  v8i g1;
  g1[0] = (int)(1u << 16);                              // wg_mask=0, data_size=1 (2B)
  g1[1] = (int)((DIM & 0xFFFFu) << 16);                 // tensor_dim0[15:0] @ bits63:48
  g1[2] = (int)((DIM >> 16) | ((rows & 0xFFFFu) << 16)); // tdim0 hi | tensor_dim1 lo
  g1[3] = (int)((rows >> 16) | ((unsigned)DIM << 16));  // tensor_dim1 hi | tile_dim0=256
  g1[4] = (int)(rows & 0xFFFFu);                        // tile_dim1=rows, tile_dim2=0
  g1[5] = (int)DIM;                                     // tensor_dim0_stride = 256
  g1[6] = 0;                                            // stride hi | dim1_stride lo
  g1[7] = 0;
  const v4i z4 = {0, 0, 0, 0};
  const v8i z8 = {0, 0, 0, 0, 0, 0, 0, 0};
  __builtin_amdgcn_tensor_load_to_lds(g0, g1, z4, z4, z8, 0);
}

// ---------------------------------------------------------------------------
// WMMA GEMM:  out[b,n] (+)= sum_k in[b,k] * W[n,k]  (+ bias[n])
//   in : BATCH x DIM f16, W: DIM x DIM f16 row-major [n][k], out: f32.
// Block tile 64(M) x 64(N), K = 256 fully staged in LDS by TDM.
// 256 threads = 8 waves, each wave: 16 rows x 32 cols (2 accumulators).
// Grid: (BATCH/64, DIM/64).
// ---------------------------------------------------------------------------
__global__ __launch_bounds__(256)
void gemm_wmma_f16f32(float* __restrict__ out,
                      const _Float16* __restrict__ in,
                      const _Float16* __restrict__ W,
                      const float* __restrict__ bias,
                      int acc) {
  __shared__ __align__(64) _Float16 sA[64 * DIM];   // 32 KB activation panel
  __shared__ __align__(64) _Float16 sW[64 * DIM];   // 32 KB weight panel

  const int lane   = threadIdx.x & 31;
  const int wave   = threadIdx.x >> 5;
  const int half   = lane >> 4;       // K-chunk select / C rows 0..7 vs 8..15
  const int l16    = lane & 15;
  const int rGroup = wave >> 1;       // 0..3 -> 16-row strip
  const int cGroup = wave & 1;        // 0..1 -> 32-col strip
  const int rowBase = blockIdx.x * 64;
  const int colBase = blockIdx.y * 64;

  // --- DMA both panels into LDS (wave 0 only), then sync the workgroup. ---
  if (wave == 0) {
    tdm_load_panel((unsigned)(uintptr_t)sA, in + (size_t)rowBase * DIM, 64u);
    tdm_load_panel((unsigned)(uintptr_t)sW, W + (size_t)colBase * DIM, 64u);
    __builtin_amdgcn_s_wait_tensorcnt(0);
  }
  __syncthreads();

  // --- accumulator init (optional C load + bias) ---
  v8f c[2];
#pragma unroll
  for (int t = 0; t < 2; ++t) {
    const int col = colBase + cGroup * 32 + t * 16 + l16;
    const float bv = bias ? bias[col] : 0.0f;
    if (acc) {
#pragma unroll
      for (int r = 0; r < 8; ++r)
        c[t][r] = out[(size_t)(rowBase + rGroup * 16 + half * 8 + r) * DIM + col] + bv;
    } else {
#pragma unroll
      for (int r = 0; r < 8; ++r) c[t][r] = bv;
    }
  }

  // --- K loop entirely out of LDS ---
  const _Float16* aRow = sA + (rGroup * 16 + l16) * DIM;
#pragma unroll
  for (int kb = 0; kb < DIM; kb += 32) {
    const v16h a = *(const v16h*)(aRow + kb + half * 16);

#pragma unroll
    for (int t = 0; t < 2; ++t) {
      const v16h bf = *(const v16h*)(sW + (cGroup * 32 + t * 16 + l16) * DIM
                                        + kb + half * 16);
      c[t] = __builtin_amdgcn_wmma_f32_16x16x32_f16(
          false, a, false, bf, (short)0, c[t], false, false);
    }
  }

#pragma unroll
  for (int t = 0; t < 2; ++t) {
    const int col = colBase + cGroup * 32 + t * 16 + l16;
#pragma unroll
    for (int r = 0; r < 8; ++r)
      out[(size_t)(rowBase + rGroup * 16 + half * 8 + r) * DIM + col] = c[t][r];
  }
}

// ---------------------------------------------------------------------------
// Elementwise kernels
// ---------------------------------------------------------------------------
__global__ void conv_w_f16(const float* __restrict__ src, _Float16* __restrict__ dst, int n) {
  int i = blockIdx.x * blockDim.x + threadIdx.x;
  if (i < n) dst[i] = (_Float16)src[i];
}

__global__ void ew_tanh_seed(const float* __restrict__ x, float* __restrict__ z0,
                             float* __restrict__ z1, _Float16* __restrict__ Th, int n) {
  int i = blockIdx.x * blockDim.x + threadIdx.x;
  if (i >= n) return;
  const float v = x[i];
  z0[i] = v;
  z1[i] = v;
  Th[i] = (_Float16)tanhf(v);
}

// Generic fused elementwise:
//   if (zout): zout = zi0 + zi1? + zi2?
//   if (uf):   uf = f16( t1*(A1?+A2?) + t2*(M1*M2? + M3*M4? + M5*M6?) + t3*(N1*N2*N3)? )
// with t=tanh(x), t1=1-t^2, t2=-2*t*t1, t3=-2*t1^2-2*t*t2.
__global__ void ew_fused(const float* __restrict__ x,
                         float* __restrict__ zout, const float* __restrict__ zi0,
                         const float* __restrict__ zi1, const float* __restrict__ zi2,
                         _Float16* __restrict__ uf,
                         const float* __restrict__ A1, const float* __restrict__ A2,
                         const float* __restrict__ M1, const float* __restrict__ M2,
                         const float* __restrict__ M3, const float* __restrict__ M4,
                         const float* __restrict__ M5, const float* __restrict__ M6,
                         const float* __restrict__ N1, const float* __restrict__ N2,
                         const float* __restrict__ N3, int n) {
  int i = blockIdx.x * blockDim.x + threadIdx.x;
  if (i >= n) return;
  if (zout) {
    float z = zi0[i];
    if (zi1) z += zi1[i];
    if (zi2) z += zi2[i];
    zout[i] = z;
  }
  if (uf) {
    const float t  = tanhf(x[i]);
    const float t1 = 1.0f - t * t;
    const float t2 = -2.0f * t * t1;
    float s = 0.0f;
    {
      float a = 0.0f; bool ha = false;
      if (A1) { a += A1[i]; ha = true; }
      if (A2) { a += A2[i]; ha = true; }
      if (ha) s += t1 * a;
    }
    {
      float m = 0.0f; bool hm = false;
      if (M1) { m += M1[i] * M2[i]; hm = true; }
      if (M3) { m += M3[i] * M4[i]; hm = true; }
      if (M5) { m += M5[i] * M6[i]; hm = true; }
      if (hm) s += t2 * m;
    }
    if (N1) {
      const float t3 = -2.0f * t1 * t1 - 2.0f * t * t2;
      s += t3 * (N1[i] * N2[i] * N3[i]);
    }
    uf[i] = (_Float16)s;
  }
}

// ---------------------------------------------------------------------------
// Host side
// ---------------------------------------------------------------------------
static inline void launch_gemm(float* out, const _Float16* in, const _Float16* W,
                               const float* bias, int acc, hipStream_t s) {
  dim3 grid(BATCH / 64, DIM / 64);
  gemm_wmma_f16f32<<<grid, 256, 0, s>>>(out, in, W, bias, acc);
}

extern "C" void kernel_launch(void* const* d_in, const int* in_sizes, int n_in,
                              void* d_out, int out_size, void* d_ws, size_t ws_size,
                              hipStream_t stream) {
  (void)in_sizes; (void)n_in; (void)out_size; (void)ws_size;

  const float* x  = (const float*)d_in[0];          // [BATCH, DIM]
  const float* Ws = (const float*)d_in[1];          // [4, DIM, DIM]
  const float* bs = (const float*)d_in[2];          // [4, DIM]

  float* zb = (float*)d_out;                        // z0..z4, each BD floats
  float* z0 = zb;
  float* z1 = zb + (size_t)BD;
  float* z2 = zb + (size_t)2 * BD;
  float* z3 = zb + (size_t)3 * BD;
  float* z4 = zb + (size_t)4 * BD;

  // Workspace layout
  char* ws = (char*)d_ws;
  float* P2 = (float*)(ws + (size_t)0 * BD * 4);
  float* P3 = (float*)(ws + (size_t)1 * BD * 4);
  float* P4 = (float*)(ws + (size_t)2 * BD * 4);
  float* B2 = (float*)(ws + (size_t)3 * BD * 4);
  float* E2 = (float*)(ws + (size_t)4 * BD * 4);
  float* S1 = (float*)(ws + (size_t)5 * BD * 4);   // w, later R2
  float* S2 = (float*)(ws + (size_t)6 * BD * 4);   // H2
  _Float16* Th = (_Float16*)(ws + (size_t)7 * BD * 4);
  _Float16* uf = (_Float16*)(ws + (size_t)7 * BD * 4 + (size_t)BD * 2);
  _Float16* Wh = (_Float16*)(ws + (size_t)7 * BD * 4 + (size_t)2 * BD * 2);

  const _Float16* W1 = Wh;
  const _Float16* W2 = Wh + (size_t)1 * DIM * DIM;
  const _Float16* W3 = Wh + (size_t)2 * DIM * DIM;
  const float* b1 = bs;
  const float* b2 = bs + DIM;
  const float* b4 = bs + 3 * DIM;

  const int n = BD;
  const int nb = (n + 255) / 256;
  const float* F0 = nullptr; float* Z0 = nullptr;

  // 0) weights -> f16
  conv_w_f16<<<(4 * DIM * DIM + 255) / 256, 256, 0, stream>>>(Ws, Wh, 4 * DIM * DIM);
  // 1) z0 = x; seed z1 = x; Th = f16(tanh(x))
  ew_tanh_seed<<<nb, 256, 0, stream>>>(x, z0, z1, Th, n);

  // 2) phi GEMMs.  z1 += W1*Th + b1 ; Pk = Wk*Th + bk
  launch_gemm(z1, Th, W1, b1, 1, stream);
  launch_gemm(P2, Th, W2, b2, 0, stream);
  launch_gemm(P3, Th, W3, bs + 2 * DIM, 0, stream);
  launch_gemm(P4, Th, Wh + (size_t)3 * DIM * DIM, b4, 0, stream);

  // 3) z2 = z1 + P2 + W1*(t1.P2)
  ew_fused<<<nb, 256, 0, stream>>>(x, z2, z1, P2, F0, uf,
                                   P2, F0, F0, F0, F0, F0, F0, F0, F0, F0, F0, n);
  launch_gemm(z2, uf, W1, nullptr, 1, stream);

  // 4) z3 = z2 + P3 + W1*u3 + B2 + W1*(t2.P3.P2 + t1.B2),  u3 = t1.P3, B2 = W2*u3
  ew_fused<<<nb, 256, 0, stream>>>(x, z3, z2, P3, F0, uf,
                                   P3, F0, F0, F0, F0, F0, F0, F0, F0, F0, F0, n);
  launch_gemm(B2, uf, W2, nullptr, 0, stream);
  launch_gemm(z3, uf, W1, nullptr, 1, stream);
  ew_fused<<<nb, 256, 0, stream>>>(x, z3, z3, B2, F0, uf,
                                   B2, F0, P3, P2, F0, F0, F0, F0, F0, F0, F0, n);
  launch_gemm(z3, uf, W1, nullptr, 1, stream);

  // 5) z4 = z3 + Dz2[P4] + Dz2[w] + D2z2[P3,P4]
  ew_fused<<<nb, 256, 0, stream>>>(x, z4, z3, P4, F0, uf,
                                   P4, F0, F0, F0, F0, F0, F0, F0, F0, F0, F0, n);
  launch_gemm(E2, uf, W2, nullptr, 0, stream);
  launch_gemm(S1, uf, W3, nullptr, 0, stream);
  ew_fused<<<nb, 256, 0, stream>>>(x, z4, z4, E2, S1, uf,
                                   P4, E2, P4, P2, F0, F0, F0, F0, F0, F0, F0, n);
  launch_gemm(z4, uf, W1, nullptr, 1, stream);
  ew_fused<<<nb, 256, 0, stream>>>(x, Z0, F0, F0, F0, uf,
                                   S1, F0, F0, F0, F0, F0, F0, F0, F0, F0, F0, n);
  launch_gemm(S2, uf, W2, nullptr, 0, stream);
  ew_fused<<<nb, 256, 0, stream>>>(x, z4, z4, S2, F0, uf,
                                   S1, S2, S1, P2, F0, F0, F0, F0, F0, F0, F0, n);
  launch_gemm(z4, uf, W1, nullptr, 1, stream);
  ew_fused<<<nb, 256, 0, stream>>>(x, Z0, F0, F0, F0, uf,
                                   F0, F0, P3, P4, F0, F0, F0, F0, F0, F0, F0, n);
  launch_gemm(S1, uf, W2, nullptr, 0, stream);
  ew_fused<<<nb, 256, 0, stream>>>(x, z4, z4, S1, F0, uf,
                                   S1, F0, P3, P4, P4, B2, P3, E2, P3, P4, P2, n);
  launch_gemm(z4, uf, W1, nullptr, 1, stream);
}
